// GaussianSplatting2D_89326729822767
// MI455X (gfx1250) — matrix operand order repaired and checked by reading, not compiled, not run on hardware
//
#include <hip/hip_runtime.h>

typedef __attribute__((ext_vector_type(16))) _Float16 v16h;
typedef __attribute__((ext_vector_type(8)))  float    v8f;

#define N_POINTS 65536
#define N_PRIMS  1024
#define N_CH     3
#define WAVES_PER_BLOCK 8
#define BLOCK_THREADS   (WAVES_PER_BLOCK * 32)   // wave32: 8 waves
#define POINTS_PER_BLOCK (WAVES_PER_BLOCK * 16)  // 16 points per wave (one WMMA N-tile)
#define KSTEPS (N_PRIMS / 32)                    // 32 WMMA K-steps of 32 prims

__global__ __launch_bounds__(BLOCK_THREADS)
void splat2d_wmma_kernel(const float* __restrict__ x,
                         const float* __restrict__ colors,
                         const float* __restrict__ means,
                         const float* __restrict__ rotations,
                         const float* __restrict__ scales,
                         float* __restrict__ out)
{
    // Per-prim covariance params (computed once per block)
    __shared__ float4 s_cov[N_PRIMS];   // {cov00, cov01, cov11, 0}
    __shared__ float2 s_mean[N_PRIMS];  // {mx, my}
    // colors^T pre-packed into WMMA A-fragments (16-bit A, 16x32), contiguous k order (same as B):
    //   lane<16 : M=lane,    halves 0..15 -> K=0..15
    //   lane>=16: M=lane-16, halves 0..15 -> K=16..31
    __shared__ v16h s_afrag[KSTEPS][32];

    const int tid = threadIdx.x;

    // ---- Phase 1: covariance from rotation/scale (1024 prims, 4 iters/thread)
    for (int p = tid; p < N_PRIMS; p += BLOCK_THREADS) {
        const float rot = rotations[p];
        const float sx  = __expf(scales[2 * p + 0]);
        const float sy  = __expf(scales[2 * p + 1]);
        const float c   = __cosf(rot);
        const float s   = __sinf(rot);
        const float a =  sx * c;
        const float b = -sy * s;
        const float d =  sx * s;
        const float e =  sy * c;
        s_cov[p]  = make_float4(a * a + b * b, a * d + b * e, d * d + e * e, 0.f);
        s_mean[p] = make_float2(means[2 * p + 0], means[2 * p + 1]);
    }

    // ---- Phase 2: pre-pack colors^T into A-fragments (f16). M=channel (3 of 16 used).
    {
        _Float16* ap = (_Float16*)&s_afrag[0][0];
        for (int t = tid; t < KSTEPS * 32 * 16; t += BLOCK_THREADS) {
            const int kstep = t >> 9;          // /(32*16)
            const int lane  = (t >> 4) & 31;
            const int h     = t & 15;
            const int ch    = lane & 15;       // matrix row M = channel
            const int kk    = h + ((lane & 16) ? 16 : 0);
            const int k     = kstep * 32 + kk; // prim index
            const float v   = (ch < N_CH) ? colors[k * N_CH + ch] : 0.f;
            ap[t] = (_Float16)v;
        }
    }

    __syncthreads();

    // ---- Phase 3: per-wave 16-point tile, full K=1024 sweep
    const int wave    = tid >> 5;
    const int lane    = tid & 31;
    const int mypoint = blockIdx.x * POINTS_PER_BLOCK + wave * 16 + (lane & 15);
    const float px = x[2 * mypoint + 0];
    const float py = x[2 * mypoint + 1];
    // B-fragment (w^T, 32x16): lanes 0-15 own K=base+0..15, lanes 16-31 own K=base+16..31
    const int khalf = (lane & 16);

    v8f acc = {};  // D/C 16x16 f32 accumulator (channels x points)

    for (int s = 0; s < KSTEPS; ++s) {
        const int kbase = s * 32 + khalf;
        v16h wfrag;
#pragma unroll
        for (int h = 0; h < 16; ++h) {
            const float4 cv = s_cov[kbase + h];   // broadcast within each half-wave
            const float2 mn = s_mean[kbase + h];
            const float dx = px - mn.x;
            const float dy = py - mn.y;
            const float q  = cv.x * dx * dx + 2.f * cv.y * (dx * dy) + cv.z * dy * dy;
            // exp(-0.5*q) = exp2(-0.5*log2(e)*q) -> v_exp_f32
            wfrag[h] = (_Float16)__builtin_amdgcn_exp2f(-0.72134752044448169f * q);
        }
        const v16h afrag = s_afrag[s][lane];
        // D = colors^T (A) x w^T (B) + acc ; emits v_wmma_f32_16x16x32_f16
        acc = __builtin_amdgcn_wmma_f32_16x16x32_f16(
            /*neg_a=*/false, afrag, /*neg_b=*/false, wfrag,
            /*c_mod=*/(short)0, acc, /*reuse_a=*/false, /*reuse_b=*/false);
    }

    // ---- Phase 4: store. D layout: VGPR r, lanes 0-15 -> M=r, N=lane.
    // Channels 0..2 live in acc[0..2] of lanes 0..15; lanes 16-31 hold M=8..15 (all zero).
    if (lane < 16) {
        out[mypoint * N_CH + 0] = acc[0];
        out[mypoint * N_CH + 1] = acc[1];
        out[mypoint * N_CH + 2] = acc[2];
    }
}

extern "C" void kernel_launch(void* const* d_in, const int* in_sizes, int n_in,
                              void* d_out, int out_size, void* d_ws, size_t ws_size,
                              hipStream_t stream) {
    (void)in_sizes; (void)n_in; (void)out_size; (void)d_ws; (void)ws_size;
    const float* x         = (const float*)d_in[0];
    const float* colors    = (const float*)d_in[1];
    const float* means     = (const float*)d_in[2];
    const float* rotations = (const float*)d_in[3];
    const float* scales    = (const float*)d_in[4];
    float* out             = (float*)d_out;

    dim3 grid(N_POINTS / POINTS_PER_BLOCK);  // 512 blocks
    dim3 block(BLOCK_THREADS);               // 256 threads = 8 wave32
    splat2d_wmma_kernel<<<grid, block, 0, stream>>>(x, colors, means, rotations, scales, out);
}
